// MetaGAT_35158602285295
// MI455X (gfx1250) — matrix operand order repaired and checked by reading, not compiled, not run on hardware
//
#include <hip/hip_runtime.h>

// MetaGAT on gfx1250 (MI455X, wave32). Two stacked GATConv layers over two
// metagraphs. GEMMs via v_wmma_f32_16x16x32_f16 fed by ds_load_b128; edge
// softmax via fused unnormalized exp-scatter (single edge pass) + per-node
// normalization.

typedef __attribute__((ext_vector_type(16))) _Float16 v16h;
typedef __attribute__((ext_vector_type(8)))  _Float16 v8h;
typedef __attribute__((ext_vector_type(4)))  _Float16 v4h;
typedef __attribute__((ext_vector_type(8)))  float    v8f;
typedef __attribute__((ext_vector_type(4)))  float    v4f;

#define SLOPE 0.2f

// Monotonic float<->uint encoding so unsigned atomicMax == float max and
// memset-to-0 is a valid "-inf" init (all real encodings are > 0).
__device__ __forceinline__ unsigned fenc(float f) {
  unsigned u = __float_as_uint(f);
  return (u & 0x80000000u) ? ~u : (u | 0x80000000u);
}
__device__ __forceinline__ float fdec(unsigned k) {
  return (k & 0x80000000u) ? __uint_as_float(k & 0x7FFFFFFFu)
                           : __uint_as_float(~k);
}

__device__ __forceinline__ v16h cat8(v8h a, v8h b) {
  return __builtin_shufflevector(a, b, 0, 1, 2, 3, 4, 5, 6, 7, 8, 9, 10, 11,
                                 12, 13, 14, 15);
}

// ---------------------------------------------------------------------------
// WMMA GEMM: C[n x 64] = A[n x K] @ W[K x 64], f32 in/out, f16 multiply,
// f32 accumulate. Block = 256 threads = 8 waves; each wave owns a 16-row
// strip, block covers 128 rows. K in {128, 64}.
//
// LDS layouts:
//   lsA  : row-major [128][K] f16. Per the 16x32 A layout, each lane's
//          fragment = two contiguous 8-half runs -> 2x ds_load_b128.
//   lsWf : fragment-major [K/32][4 ntiles][32 lanes][16 elems] f16, so each
//          lane's B fragment is 32 contiguous bytes -> 2x ds_load_b128.
//          (element e of lane L maps to W[kt*32 + (L>>4)*16 + e][nt*16+(L&15)])
// ---------------------------------------------------------------------------
template <int K>
__global__ __launch_bounds__(256) void gat_gemm_kernel(
    const float* __restrict__ A, const float* __restrict__ W,
    float* __restrict__ Cout, int nRows) {
  __shared__ _Float16 lsA[128 * K];
  __shared__ _Float16 lsWf[K * 64];

  const int tid = threadIdx.x;
  const int rowBase = blockIdx.x * 128;

  // Stage W in fragment-major order (scattered reads; W is tiny & L2-hot).
  for (int i = tid; i < K * 64; i += 256) {
    int e  = i & 15;
    int ln = (i >> 4) & 31;
    int t2 = i >> 9;  // kt*4 + nt
    int k  = (t2 >> 2) * 32 + (ln >> 4) * 16 + e;
    int nn = (t2 & 3) * 16 + (ln & 15);
    lsWf[i] = (_Float16)W[k * 64 + nn];
  }
  // Stage A rows: coalesced float4 global loads -> f16x4 LDS stores.
  constexpr int KV = K / 4;
  for (int i = tid; i < 128 * KV; i += 256) {
    int r = i / KV, kq = i - r * KV;
    int gr = rowBase + r;
    v4f f = (gr < nRows) ? *(const v4f*)(A + (size_t)gr * K + kq * 4)
                         : (v4f){0.f, 0.f, 0.f, 0.f};
    *(v4h*)(lsA + r * K + kq * 4) = __builtin_convertvector(f, v4h);
  }
  __syncthreads();

  const int wave = tid >> 5;
  const int lane = tid & 31;
  const int half = lane >> 4;       // lane group (K-split for A & B)
  const int mr   = lane & 15;       // A: row in tile ; B/C: column n
  const int lr   = wave * 16 + mr;  // local A row this lane sources

  v8f acc[4] = {};  // 4 n-tiles of 16 columns

#pragma unroll
  for (int kt = 0; kt < K / 32; ++kt) {
    // A fragment: elems 0..15 -> K=kt*32+16*half+e.
    const _Float16* pa = lsA + lr * K + kt * 32 + half * 16;
    v16h afrag = cat8(*(const v8h*)pa, *(const v8h*)(pa + 8));

#pragma unroll
    for (int nt = 0; nt < 4; ++nt) {
      const _Float16* pb = lsWf + (((kt * 4 + nt) * 32 + lane) << 4);
      v16h bfrag = cat8(*(const v8h*)pb, *(const v8h*)(pb + 8));
      acc[nt] = __builtin_amdgcn_wmma_f32_16x16x32_f16(
          false, afrag, false, bfrag, (short)0, acc[nt], false, false);
    }
  }

  // C layout: vgpr r -> row m = r + 8*half ; column = lane&15.
  const int rowTile = rowBase + wave * 16 + half * 8;
  if (rowBase + 128 <= nRows) {  // uniform fast path: no predication
#pragma unroll
    for (int nt = 0; nt < 4; ++nt)
#pragma unroll
      for (int r = 0; r < 8; ++r)
        Cout[(size_t)(rowTile + r) * 64 + nt * 16 + mr] = acc[nt][r];
  } else {
#pragma unroll
    for (int nt = 0; nt < 4; ++nt)
#pragma unroll
      for (int r = 0; r < 8; ++r)
        if (rowTile + r < nRows)
          Cout[(size_t)(rowTile + r) * 64 + nt * 16 + mr] = acc[nt][r];
  }
}

// ---------------------------------------------------------------------------
// al_s[n,h] = sum_c xt[n,h,c]*a_src[h,c] ; al_d likewise. H*C == 64.
// ---------------------------------------------------------------------------
__global__ void attn_coef_kernel(const float* __restrict__ xt,
                                 const float* __restrict__ a_src,
                                 const float* __restrict__ a_dst,
                                 float* __restrict__ al_s,
                                 float* __restrict__ al_d, int n, int H, int C) {
  int t = blockIdx.x * blockDim.x + threadIdx.x;
  if (t >= n * H) return;
  int node = t / H, h = t - node * H;
  const float* row = xt + (size_t)node * 64 + h * C;
  float ss = 0.f, sd = 0.f;
  for (int c = 0; c < C; ++c) {
    float v = row[c];
    ss += v * a_src[h * C + c];
    sd += v * a_dst[h * C + c];
  }
  al_s[t] = ss;
  al_d[t] = sd;
}

// ---------------------------------------------------------------------------
// Segment max over destination: one thread per (edge, head); self-loops are
// the trailing n virtual edges. emaxKey memset to 0 == "-inf".
// ---------------------------------------------------------------------------
__global__ void edge_max_kernel(const int* __restrict__ src,
                                const int* __restrict__ dst, int E, int n,
                                int H, const float* __restrict__ al_s,
                                const float* __restrict__ al_d,
                                unsigned* __restrict__ emaxKey) {
  int t = blockIdx.x * blockDim.x + threadIdx.x;
  if (t >= (E + n) * H) return;
  int e = t / H, h = t - e * H;
  int s = (e < E) ? src[e] : (e - E);
  int d = (e < E) ? dst[e] : (e - E);
  float v = al_s[s * H + h] + al_d[d * H + h];
  v = (v > 0.f) ? v : SLOPE * v;
  atomicMax(&emaxKey[d * H + h], fenc(v));
}

// ---------------------------------------------------------------------------
// Fused edge pass: one wave32 per edge. Scatters exp-weighted (unnormalized)
// source features + accumulates the softmax denominator in the same pass.
// All per-node arrays are L2-resident (12.8 MB << 192 MB L2).
// ---------------------------------------------------------------------------
template <int H>
__global__ __launch_bounds__(256) void edge_scatter_kernel(
    const int* __restrict__ src, const int* __restrict__ dst, int E, int n,
    const float* __restrict__ al_s, const float* __restrict__ al_d,
    const unsigned* __restrict__ emaxKey, const float* __restrict__ xt,
    float* __restrict__ denom, float* __restrict__ acc) {
  const int wv = (blockIdx.x * blockDim.x + threadIdx.x) >> 5;
  const int lane = threadIdx.x & 31;
  if (wv >= E + n) return;
  const int s = (wv < E) ? src[wv] : (wv - E);
  const int d = (wv < E) ? dst[wv] : (wv - E);
  constexpr int C = 64 / H;

  float w[H];
#pragma unroll
  for (int h = 0; h < H; ++h) {
    float v = al_s[s * H + h] + al_d[d * H + h];
    v = (v > 0.f) ? v : SLOPE * v;
    w[h] = __expf(v - fdec(emaxKey[d * H + h]));
  }
  if (lane < H) atomicAdd(&denom[d * H + lane], w[lane]);
#pragma unroll
  for (int j = lane; j < 64; j += 32)
    atomicAdd(&acc[(size_t)d * 64 + j], xt[(size_t)s * 64 + j] * w[j / C]);
}

// ---------------------------------------------------------------------------
// Layer-1 finalize: normalize, add bias, ELU.
// ---------------------------------------------------------------------------
__global__ void finalize_elu_kernel(const float* __restrict__ acc,
                                    const float* __restrict__ denom,
                                    const float* __restrict__ bias,
                                    float* __restrict__ out, int n, int H) {
  int t = blockIdx.x * blockDim.x + threadIdx.x;
  if (t >= n * 64) return;
  int node = t >> 6, j = t & 63;
  int C = 64 / H;
  float v = acc[t] / denom[node * H + j / C] + bias[j];
  out[t] = (v > 0.f) ? v : expm1f(v);
}

// ---------------------------------------------------------------------------
// Layer-2 finalize: normalize (H==1), add bias, log_softmax over 64 feats.
// One wave per node; 2 elems/lane; shfl_xor reductions (wave32).
// ---------------------------------------------------------------------------
__global__ __launch_bounds__(256) void finalize_logsoftmax_kernel(
    const float* __restrict__ acc, const float* __restrict__ denom,
    const float* __restrict__ bias, float* __restrict__ out, int n) {
  const int wv = (blockIdx.x * blockDim.x + threadIdx.x) >> 5;
  const int lane = threadIdx.x & 31;
  if (wv >= n) return;
  const float inv = 1.f / denom[wv];
  float v0 = acc[(size_t)wv * 64 + lane] * inv + bias[lane];
  float v1 = acc[(size_t)wv * 64 + lane + 32] * inv + bias[lane + 32];
  float mx = fmaxf(v0, v1);
  for (int off = 16; off; off >>= 1) mx = fmaxf(mx, __shfl_xor(mx, off, 32));
  float se = __expf(v0 - mx) + __expf(v1 - mx);
  for (int off = 16; off; off >>= 1) se += __shfl_xor(se, off, 32);
  float lse = mx + __logf(se);
  out[(size_t)wv * 64 + lane] = v0 - lse;
  out[(size_t)wv * 64 + lane + 32] = v1 - lse;
}

// ---------------------------------------------------------------------------
static inline int cdiv(long long a, long long b) { return (int)((a + b - 1) / b); }

extern "C" void kernel_launch(void* const* d_in, const int* in_sizes, int n_in,
                              void* d_out, int out_size, void* d_ws,
                              size_t ws_size, hipStream_t stream) {
  (void)n_in; (void)out_size; (void)ws_size;
  const float* x = (const float*)d_in[0];
  const int* ei[2] = {(const int*)d_in[1], (const int*)d_in[2]};
  const int n = in_sizes[0] / 128;  // F_DIM = 128
  const int E = in_sizes[1] / 2;

  // Workspace carve-out (~41.6 MB for N=50000).
  char* ws = (char*)d_ws;
  size_t off = 0;
  auto carve = [&](size_t bytes) -> void* {
    void* p = ws + off;
    off += (bytes + 255) & ~(size_t)255;
    return p;
  };
  float*    xt    = (float*)carve((size_t)n * 64 * 4);  // GEMM output
  float*    acc   = (float*)carve((size_t)n * 64 * 4);  // scatter accumulator
  float*    hbuf  = (float*)carve((size_t)n * 64 * 4);  // layer-1 activations
  float*    al_s  = (float*)carve((size_t)n * 4 * 4);
  float*    al_d  = (float*)carve((size_t)n * 4 * 4);
  unsigned* emax  = (unsigned*)carve((size_t)n * 4 * 4);
  float*    denom = (float*)carve((size_t)n * 4 * 4);

  for (int m = 0; m < 2; ++m) {
    const int base = 3 + m * 8;
    const float* W1  = (const float*)d_in[base + 0];
    const float* as1 = (const float*)d_in[base + 1];
    const float* ad1 = (const float*)d_in[base + 2];
    const float* b1  = (const float*)d_in[base + 3];
    const float* W2  = (const float*)d_in[base + 4];
    const float* as2 = (const float*)d_in[base + 5];
    const float* ad2 = (const float*)d_in[base + 6];
    const float* b2  = (const float*)d_in[base + 7];
    const int* srcI = ei[m];
    const int* dstI = ei[m] + E;

    // ---- layer 1: GATConv(128 -> 4 heads x 16), ELU ----
    gat_gemm_kernel<128><<<cdiv(n, 128), 256, 0, stream>>>(x, W1, xt, n);
    attn_coef_kernel<<<cdiv((long long)n * 4, 256), 256, 0, stream>>>(
        xt, as1, ad1, al_s, al_d, n, 4, 16);
    hipMemsetAsync(emax, 0, (size_t)n * 4 * 4, stream);
    hipMemsetAsync(denom, 0, (size_t)n * 4 * 4, stream);
    hipMemsetAsync(acc, 0, (size_t)n * 64 * 4, stream);
    edge_max_kernel<<<cdiv((long long)(E + n) * 4, 256), 256, 0, stream>>>(
        srcI, dstI, E, n, 4, al_s, al_d, emax);
    edge_scatter_kernel<4><<<cdiv((long long)(E + n) * 32, 256), 256, 0,
                             stream>>>(srcI, dstI, E, n, al_s, al_d, emax, xt,
                                       denom, acc);
    finalize_elu_kernel<<<cdiv((long long)n * 64, 256), 256, 0, stream>>>(
        acc, denom, b1, hbuf, n, 4);

    // ---- layer 2: GATConv(64 -> 64, 1 head), log_softmax ----
    gat_gemm_kernel<64><<<cdiv(n, 128), 256, 0, stream>>>(hbuf, W2, xt, n);
    attn_coef_kernel<<<cdiv(n, 256), 256, 0, stream>>>(xt, as2, ad2, al_s,
                                                       al_d, n, 1, 64);
    hipMemsetAsync(emax, 0, (size_t)n * 4, stream);
    hipMemsetAsync(denom, 0, (size_t)n * 4, stream);
    hipMemsetAsync(acc, 0, (size_t)n * 64 * 4, stream);
    edge_max_kernel<<<cdiv((long long)(E + n), 256), 256, 0, stream>>>(
        srcI, dstI, E, n, 1, al_s, al_d, emax);
    edge_scatter_kernel<1><<<cdiv((long long)(E + n) * 32, 256), 256, 0,
                             stream>>>(srcI, dstI, E, n, al_s, al_d, emax, xt,
                                       denom, acc);
    finalize_logsoftmax_kernel<<<cdiv((long long)n * 32, 256), 256, 0,
                                 stream>>>(acc, denom, b2,
                                           (float*)d_out + (size_t)m * n * 64,
                                           n);
  }
}
